// GIN_46377056862924
// MI455X (gfx1250) — matrix operand order repaired and checked by reading, not compiled, not run on hardware
//
#include <hip/hip_runtime.h>

// ---------------- types ----------------
typedef __attribute__((ext_vector_type(16))) _Float16 v16h;
typedef __attribute__((ext_vector_type(8)))  _Float16 v8h;
typedef __attribute__((ext_vector_type(8)))  float    v8f;

union V16U { v16h v; v8h h[2]; };

#define D_IN  128
#define D_H   256
#define D_OUT 64
// fragment layout: 32 lanes x 16 halves = 512 halves = 1 KB per 32x16 B-tile
#define FRAG_HALVES 512
#define W1_FRAGS (16*4)          // 16 n-tiles x 4 k-frags
#define W2_FRAGS (16*8)          // 16 n-tiles x 8 k-frags
#define WC_FRAGS (4*8)           //  4 n-tiles x 8 k-frags
#define W2_OFF (W1_FRAGS*FRAG_HALVES)                 // 32768 halves
#define WC_OFF ((W1_FRAGS+W2_FRAGS)*FRAG_HALVES)      // 98304 halves
#define N_WELEM (D_IN*D_H + D_H*D_H + D_H*D_OUT)      // 114688

#define WMMA_F16(A,B,C) \
  __builtin_amdgcn_wmma_f32_16x16x32_f16(false,(A),false,(B),(short)0,(C),false,false)

// ---------------- CSR build: histogram of destination degrees ----------------
__global__ __launch_bounds__(256)
void gin_hist(const long long* __restrict__ ei, int* __restrict__ counts, int n_edges) {
  int e = blockIdx.x * 256 + threadIdx.x;
  if (e >= n_edges) return;
  atomicAdd(counts + (int)ei[n_edges + e], 1);
}

// ---------------- CSR build: single-block exclusive scan (n ~ 50k) ----------------
__global__ __launch_bounds__(1024)
void gin_scan(const int* __restrict__ counts, int* __restrict__ offs, int n) {
  __shared__ int buf[1024];
  __shared__ int carry;
  if (threadIdx.x == 0) carry = 0;
  __syncthreads();
  for (int base = 0; base < n; base += 1024) {
    int i = base + (int)threadIdx.x;
    int v = (i < n) ? counts[i] : 0;
    buf[threadIdx.x] = v;
    __syncthreads();
    #pragma unroll
    for (int off = 1; off < 1024; off <<= 1) {
      int t = (threadIdx.x >= (unsigned)off) ? buf[threadIdx.x - off] : 0;
      __syncthreads();
      buf[threadIdx.x] += t;
      __syncthreads();
    }
    if (i < n) offs[i] = carry + buf[threadIdx.x] - v;   // exclusive
    __syncthreads();
    if (threadIdx.x == 0) carry += buf[1023];
    __syncthreads();
  }
  if (threadIdx.x == 0) offs[n] = carry;
}

// ---------------- CSR build: fill per-destination source lists ----------------
__global__ __launch_bounds__(256)
void gin_fill(const long long* __restrict__ ei, int* __restrict__ cursor,
              int* __restrict__ eidx, int n_edges) {
  int e = blockIdx.x * 256 + threadIdx.x;
  if (e >= n_edges) return;
  int src = (int)ei[e];
  int dst = (int)ei[n_edges + e];
  int pos = atomicAdd(cursor + dst, 1);
  eidx[pos] = src;
}

// ---------------- gather-sum: hbuf[i] = x[i] + sum_{j->i} x[j] ----------------
// one wave per node, lane owns 4 consecutive features; zero float atomics.
// Neighbor indices are fetched cooperatively (one per lane) and broadcast with
// __shfl, so row gathers stream back-to-back instead of chaining on index loads.
__global__ __launch_bounds__(256)
void gin_gather(const float* __restrict__ x, const int* __restrict__ offs,
                const int* __restrict__ eidx, float* __restrict__ hbuf, int n) {
  int node = blockIdx.x * 8 + (threadIdx.x >> 5);
  if (node >= n) return;
  int lane = threadIdx.x & 31;
  const float4 xv = *(const float4*)(x + (size_t)node * D_IN + lane * 4);
  float a0 = xv.x, a1 = xv.y, a2 = xv.z, a3 = xv.w;
  int p0 = offs[node], p1 = offs[node + 1];
  for (int p = p0; p < p1; p += 32) {
    int navail = p1 - p;
    int cnt = navail < 32 ? navail : 32;
    int myidx = eidx[p + (lane < cnt ? lane : 0)];    // clamped, always in-bounds
    for (int q = 0; q < cnt; ++q) {
      int s = __shfl(myidx, q, 32);
      const float4 v = *(const float4*)(x + (size_t)s * D_IN + lane * 4);
      a0 += v.x; a1 += v.y; a2 += v.z; a3 += v.w;
    }
  }
  float4 r; r.x = a0; r.y = a1; r.z = a2; r.w = a3;
  *(float4*)(hbuf + (size_t)node * D_IN + lane * 4) = r;
}

// ---------------- repack weights fp32 -> f16 B-fragment layout ----------------
// B 32x16 f16 layout (wave32): lane L holds B[K=(j<8 ? 8h+j : 16+8h+j-8)][N=L&15], h=L>=16, j=0..15.
__global__ __launch_bounds__(256)
void gin_repack(const float* __restrict__ W1, const float* __restrict__ W2,
                const float* __restrict__ Wc, _Float16* __restrict__ wf) {
  int i = blockIdx.x * 256 + threadIdx.x;
  if (i >= N_WELEM) return;
  int k, n, base, KF;
  float val;
  if (i < D_IN * D_H) {                     // W1 [128][256]
    k = i >> 8; n = i & 255; base = 0; KF = 4; val = W1[i];
  } else if (i < D_IN * D_H + D_H * D_H) {  // W2 [256][256]
    int j = i - D_IN * D_H;
    k = j >> 8; n = j & 255; base = W2_OFF; KF = 8; val = W2[j];
  } else {                                  // Wc [256][64]
    int j = i - (D_IN * D_H + D_H * D_H);
    k = j >> 6; n = j & 63; base = WC_OFF; KF = 8; val = Wc[j];
  }
  int nt = n >> 4;
  int kf = k >> 5;
  int kr = k & 31;
  int L  = (n & 15) + ((kr & 8) ? 16 : 0);
  int jj = (kr & 7) | ((kr & 16) >> 1);
  wf[base + ((size_t)(nt * KF + kf) * 32 + L) * 16 + jj] = (_Float16)val;
}

// ---------------- fused GIN MLP with WMMA ----------------
// 1 wave == 16 nodes. LDS scratch per wave for the C->A cross-lane transpose.
__global__ __launch_bounds__(128)
void gin_mlp(const float* __restrict__ hbuf, const _Float16* __restrict__ wf,
             const float* __restrict__ b1, const float* __restrict__ b2,
             const float* __restrict__ bc,
             float* __restrict__ out, int ntiles) {
  __shared__ _Float16 smem[4][2][16 * D_H];   // 64 KB: 4 waves x 2 x (16x256) f16
  const int wave = threadIdx.x >> 5;
  const int lane = threadIdx.x & 31;
  const int tile = blockIdx.x * 4 + wave;     // wave-uniform -> EXEC stays all-ones
  if (tile >= ntiles) return;
  const int col = lane & 15;                  // N in C-layout; M(row) in A-layout
  const int hi8 = (lane >> 4) << 3;           // 0 or 8
  _Float16* S1 = &smem[wave][0][0];
  _Float16* S2 = &smem[wave][1][0];

  // ---- A fragments of H  (16x128 f16, A layout per ISA) ----
  const float* hr = hbuf + (size_t)(tile * 16 + col) * D_IN;
  v16h a1[4];
  #pragma unroll
  for (int kf = 0; kf < 4; ++kf) {
    const int c0 = kf * 32 + hi8;
    #pragma unroll
    for (int q = 0; q < 8; ++q) {
      a1[kf][q]     = (_Float16)hr[c0 + q];
      a1[kf][q + 8] = (_Float16)hr[c0 + 16 + q];
    }
  }

  // ---- GEMM1: h1 = relu(H @ W1 + b1)  [16x256] ----
  #pragma unroll
  for (int nt = 0; nt < 16; ++nt) {
    float bv = b1[nt * 16 + col];
    v8f acc;
    #pragma unroll
    for (int r = 0; r < 8; ++r) acc[r] = bv;
    #pragma unroll
    for (int kf = 0; kf < 4; ++kf) {
      v16h bf = *(const v16h*)(wf + ((size_t)(nt * 4 + kf) * 32 + lane) * 16);
      acc = WMMA_F16(a1[kf], bf, acc);
    }
    #pragma unroll
    for (int r = 0; r < 8; ++r) {
      float v = acc[r] > 0.f ? acc[r] : 0.f;
      S1[(r + hi8) * D_H + nt * 16 + col] = (_Float16)v;
    }
  }

  // ---- re-load h1 as A fragments (cross-lane transpose via LDS) ----
  v16h a2[8];
  #pragma unroll
  for (int kf = 0; kf < 8; ++kf) {
    const int c0 = kf * 32 + hi8;
    V16U u;
    u.h[0] = *(const v8h*)(S1 + col * D_H + c0);
    u.h[1] = *(const v8h*)(S1 + col * D_H + c0 + 16);
    a2[kf] = u.v;
  }

  // ---- GEMM2: h2 = h1 @ W2 + b2; store relu(h2) ----
  #pragma unroll
  for (int nt = 0; nt < 16; ++nt) {
    float bv = b2[nt * 16 + col];
    v8f acc;
    #pragma unroll
    for (int r = 0; r < 8; ++r) acc[r] = bv;
    #pragma unroll
    for (int kf = 0; kf < 8; ++kf) {
      v16h bf = *(const v16h*)(wf + W2_OFF + ((size_t)(nt * 8 + kf) * 32 + lane) * 16);
      acc = WMMA_F16(a2[kf], bf, acc);
    }
    #pragma unroll
    for (int r = 0; r < 8; ++r) {
      float v = acc[r] > 0.f ? acc[r] : 0.f;  // classifier input is relu(h2)
      S2[(r + hi8) * D_H + nt * 16 + col] = (_Float16)v;
    }
  }

  // ---- re-load relu(h2) as A fragments ----
  v16h a3[8];
  #pragma unroll
  for (int kf = 0; kf < 8; ++kf) {
    const int c0 = kf * 32 + hi8;
    V16U u;
    u.h[0] = *(const v8h*)(S2 + col * D_H + c0);
    u.h[1] = *(const v8h*)(S2 + col * D_H + c0 + 16);
    a3[kf] = u.v;
  }

  // ---- GEMM3: out = relu(h2) @ Wc + bc  [16x64] ----
  #pragma unroll
  for (int nt = 0; nt < 4; ++nt) {
    float bv = bc[nt * 16 + col];
    v8f acc;
    #pragma unroll
    for (int r = 0; r < 8; ++r) acc[r] = bv;
    #pragma unroll
    for (int kf = 0; kf < 8; ++kf) {
      v16h bf = *(const v16h*)(wf + WC_OFF + ((size_t)(nt * 8 + kf) * 32 + lane) * 16);
      acc = WMMA_F16(a3[kf], bf, acc);
    }
    #pragma unroll
    for (int r = 0; r < 8; ++r)
      out[(size_t)(tile * 16 + r + hi8) * D_OUT + nt * 16 + col] = acc[r];
  }
}

// ---------------- host launcher ----------------
static inline size_t align256(size_t v) { return (v + 255) & ~(size_t)255; }

extern "C" void kernel_launch(void* const* d_in, const int* in_sizes, int n_in,
                              void* d_out, int out_size, void* d_ws, size_t ws_size,
                              hipStream_t stream) {
  (void)n_in; (void)out_size; (void)ws_size;
  const float*     x  = (const float*)d_in[0];
  const long long* ei = (const long long*)d_in[1];   // edge_index int64 [2, E]
  const float*     W1 = (const float*)d_in[2];
  const float*     b1 = (const float*)d_in[3];
  const float*     W2 = (const float*)d_in[4];
  const float*     b2 = (const float*)d_in[5];
  const float*     Wc = (const float*)d_in[6];
  const float*     bc = (const float*)d_in[7];
  float* out = (float*)d_out;

  const int N       = in_sizes[0] / D_IN;   // 50000
  const int n_edges = in_sizes[1] / 2;      // 800000

  // ---- workspace carve-up ----
  char* p = (char*)d_ws;
  float*    hbuf   = (float*)p;            p += align256((size_t)N * D_IN * sizeof(float));
  _Float16* wf     = (_Float16*)p;         p += align256((size_t)N_WELEM * sizeof(_Float16));
  int*      counts = (int*)p;              p += align256((size_t)N * sizeof(int));
  int*      offs   = (int*)p;              p += align256((size_t)(N + 1) * sizeof(int));
  int*      cursor = (int*)p;              p += align256((size_t)N * sizeof(int));
  int*      eidx   = (int*)p;              p += align256((size_t)n_edges * sizeof(int));

  const int eblocks = (n_edges + 255) / 256;

  // ---- CSR build (int atomics only) ----
  hipMemsetAsync(counts, 0, (size_t)N * sizeof(int), stream);
  gin_hist<<<eblocks, 256, 0, stream>>>(ei, counts, n_edges);
  gin_scan<<<1, 1024, 0, stream>>>(counts, offs, N);
  hipMemcpyAsync(cursor, offs, (size_t)N * sizeof(int),
                 hipMemcpyDeviceToDevice, stream);
  gin_fill<<<eblocks, 256, 0, stream>>>(ei, cursor, eidx, n_edges);

  // ---- deterministic gather-sum: hbuf = x + segment_sum(x[src], dst) ----
  gin_gather<<<(N + 7) / 8, 256, 0, stream>>>(x, offs, eidx, hbuf, N);

  // ---- weights -> f16 fragments (L2-resident, 224 KB) ----
  gin_repack<<<(N_WELEM + 255) / 256, 256, 0, stream>>>(W1, W2, Wc, wf);

  // ---- fused MLP on WMMA ----
  const int ntiles = N / 16;                // 3125
  gin_mlp<<<(ntiles + 3) / 4, 128, 0, stream>>>(hbuf, wf, b1, b2, bc, out, ntiles);
}
